// Head_3186865733867
// MI455X (gfx1250) — compile-verified, matrix operand order checked by reading, not run on hardware
//
#include <hip/hip_runtime.h>

#define B_  8
#define S_  2048
#define DM_ 1024
#define DH_ 64

typedef __attribute__((ext_vector_type(16))) __bf16 v16bf;
typedef __attribute__((ext_vector_type(8)))  float  v8f;
typedef __attribute__((ext_vector_type(4)))  int    v4i;
typedef __attribute__((address_space(1))) v4i gv4i;   // global
typedef __attribute__((address_space(3))) v4i lv4i;   // LDS

#if defined(__has_builtin)
#if __has_builtin(__builtin_amdgcn_global_load_async_to_lds_b128) && \
    __has_builtin(__builtin_amdgcn_s_wait_asynccnt)
#define USE_ASYNC_LDS 1
#endif
#endif

__device__ __forceinline__ unsigned short f2bfu(float f) {
  return __builtin_bit_cast(unsigned short, (__bf16)f);   // RNE hardware cvt
}

// Build a v16bf fragment from two 16-byte chunks (jj 0..7 and jj 8..15 are
// contiguous in K for the 16-bit WMMA A/B layouts).
__device__ __forceinline__ v16bf frag_ld(const unsigned short* p0,
                                         const unsigned short* p1) {
  struct U { uint4 lo, hi; } t;
  t.lo = *(const uint4*)p0;
  t.hi = *(const uint4*)p1;
  return __builtin_bit_cast(v16bf, t);
}

// ---------------------------------------------------------------------------
// One-shot: W [DM, DH] f32 -> W^T [DH, DM] bf16 (so proj B-fragments are
// contiguous-b128 loads). 3 x 64K elements; negligible cost, L2 resident.
// ---------------------------------------------------------------------------
__global__ void __launch_bounds__(256)
wt_kernel(const float* __restrict__ Wq, const float* __restrict__ Wk,
          const float* __restrict__ Wv, unsigned short* __restrict__ wtq,
          unsigned short* __restrict__ wtk, unsigned short* __restrict__ wtv) {
  const float* W; unsigned short* O;
  if (blockIdx.y == 0)      { W = Wq; O = wtq; }
  else if (blockIdx.y == 1) { W = Wk; O = wtk; }
  else                      { W = Wv; O = wtv; }
  const int i = blockIdx.x * 256 + threadIdx.x;   // 0 .. DH*DM-1
  const int h = i >> 10;                          // output row (head col)
  const int d = i & (DM_ - 1);                    // output col (model dim)
  O[i] = f2bfu(W[(size_t)d * DH_ + h]);
}

// ---------------------------------------------------------------------------
// Projection: [B*S, DM] f32 x [DM, DH] -> bf16 [B*S, DH] in workspace.
// grid = (1024, 3), 128 threads (4 waves); wave w owns output cols [16w,16w+16).
// A from global f32 (b128 + cvt_pk_bf16), B from bf16 W^T (2 x b128).
// ---------------------------------------------------------------------------
__global__ void __launch_bounds__(128)
proj_kernel(const float* __restrict__ q, const float* __restrict__ k,
            const float* __restrict__ v, const unsigned short* __restrict__ wtq,
            const unsigned short* __restrict__ wtk,
            const unsigned short* __restrict__ wtv,
            unsigned short* __restrict__ qh, unsigned short* __restrict__ kh,
            unsigned short* __restrict__ vh) {
  const float* X; const unsigned short* Wt; unsigned short* O;
  if (blockIdx.y == 0)      { X = q; Wt = wtq; O = qh; }
  else if (blockIdx.y == 1) { X = k; Wt = wtk; O = kh; }
  else                      { X = v; Wt = wtv; O = vh; }

  const int row0 = blockIdx.x * 16;
  const int tid  = threadIdx.x;
  const int wave = tid >> 5;
  const int lane = tid & 31;
  const int half = lane >> 4;
  const int ln   = lane & 15;

  const float*          arow = X  + (size_t)(row0 + ln) * DM_ + half * 8;
  const unsigned short* brow = Wt + (size_t)(wave * 16 + ln) * DM_ + half * 8;

  v8f acc = {};
  for (int k0 = 0; k0 < DM_; k0 += 32) {
    // A fragment: rows ln, K = k0 + (jj<8?jj:jj+8) + 8*half  -> 4 x b128 f32
    const float4 f0 = *(const float4*)(arow + k0);
    const float4 f1 = *(const float4*)(arow + k0 + 4);
    const float4 f2 = *(const float4*)(arow + k0 + 16);
    const float4 f3 = *(const float4*)(arow + k0 + 20);
    v16bf a;
    a[0]=(__bf16)f0.x; a[1]=(__bf16)f0.y; a[2]=(__bf16)f0.z; a[3]=(__bf16)f0.w;
    a[4]=(__bf16)f1.x; a[5]=(__bf16)f1.y; a[6]=(__bf16)f1.z; a[7]=(__bf16)f1.w;
    a[8]=(__bf16)f2.x; a[9]=(__bf16)f2.y; a[10]=(__bf16)f2.z; a[11]=(__bf16)f2.w;
    a[12]=(__bf16)f3.x; a[13]=(__bf16)f3.y; a[14]=(__bf16)f3.z; a[15]=(__bf16)f3.w;
    // B fragment from W^T: contiguous 2 x b128
    const v16bf b = frag_ld(brow + k0, brow + k0 + 16);
    acc = __builtin_amdgcn_wmma_f32_16x16x32_bf16(false, a, false, b, (short)0,
                                                  acc, false, false);
  }

#pragma unroll
  for (int r = 0; r < 8; ++r) {
    const int m = r + half * 8;
    O[(size_t)(row0 + m) * DH_ + wave * 16 + ln] = f2bfu(acc[r]);
  }
}

// ---------------------------------------------------------------------------
// Flash attention, causal. Block = 64 query rows (4 waves x 16 rows).
// K tile staged to LDS via async-to-LDS b128 (CDNA5 path); V tile staged
// TRANSPOSED (packed b32 stores) so P.V B-fragments are contiguous b128 reads.
// ---------------------------------------------------------------------------
__global__ void __launch_bounds__(128)
attn_kernel(const unsigned short* __restrict__ qh,
            const unsigned short* __restrict__ kh,
            const unsigned short* __restrict__ vh,
            float* __restrict__ out) {
  const int b  = blockIdx.x >> 5;      // S/64 == 32 query tiles per batch
  const int qt = blockIdx.x & 31;
  const int q0 = qt * 64;

  const int tid  = threadIdx.x;
  const int wave = tid >> 5;
  const int lane = tid & 31;
  const int half = lane >> 4;
  const int ln   = lane & 15;
  const int rowbase = q0 + wave * 16;

  __shared__ unsigned short lds_k[32 * 64];       // [key][dh] row-major
  __shared__ unsigned short lds_vt[64 * 32];      // [dh][key] transposed
  __shared__ unsigned short lds_p[4][16 * 32];    // per-wave P transpose

  // Q fragments: 16x64 per wave as two K=32 A-fragments (4 x b128 global)
  const unsigned short* qrow = qh + ((size_t)b * S_ + rowbase + ln) * DH_ + half * 8;
  v16bf aq[2];
  aq[0] = frag_ld(qrow,      qrow + 16);
  aq[1] = frag_ld(qrow + 32, qrow + 48);

  v8f o[4] = {};
  float mrow[8], lrow[8];
#pragma unroll
  for (int r = 0; r < 8; ++r) { mrow[r] = -1e30f; lrow[r] = 0.0f; }

  const int str = tid >> 2;            // K staging: key row 0..31
  const int stc = (tid & 3) << 4;      // K staging: dh col {0,16,32,48}
  const int vrp = (tid & 15) << 1;     // V staging: key rows {vrp, vrp+1}
  const int vcg = (tid >> 4) << 3;     // V staging: dh cols vcg..vcg+7

  const int ntiles = (q0 + 64) >> 5;   // causal: only keys <= q0+63
  for (int t = 0; t < ntiles; ++t) {
    const int j = t << 5;
    __syncthreads();
    {
      const unsigned short* kg = kh + ((size_t)b * S_ + j + str) * DH_ + stc;
#ifdef USE_ASYNC_LDS
      __builtin_amdgcn_global_load_async_to_lds_b128(
          (gv4i*)kg, (lv4i*)&lds_k[str * 64 + stc], 0, 0);
      __builtin_amdgcn_global_load_async_to_lds_b128(
          (gv4i*)kg, (lv4i*)&lds_k[str * 64 + stc], 16, 0);
#else
      *(uint4*)&lds_k[str * 64 + stc]     = *(const uint4*)kg;
      *(uint4*)&lds_k[str * 64 + stc + 8] = *(const uint4*)(kg + 8);
#endif
      // V: two adjacent key rows x 8 cols; pack (row, row+1) pairs -> b32
      const unsigned short* vg0 = vh + ((size_t)b * S_ + j + vrp) * DH_ + vcg;
      union { uint4 v4; unsigned short us[8]; } r0, r1;
      r0.v4 = *(const uint4*)vg0;
      r1.v4 = *(const uint4*)(vg0 + DH_);
#pragma unroll
      for (int i = 0; i < 8; ++i) {
        const unsigned int pk =
            (unsigned int)r0.us[i] | ((unsigned int)r1.us[i] << 16);
        *(unsigned int*)&lds_vt[(vcg + i) * 32 + vrp] = pk;
      }

      if (t + 1 < ntiles) {            // prefetch next K/V tile into caches
        __builtin_prefetch(kg + 32 * DH_, 0, 3);
        __builtin_prefetch(vg0 + 32 * DH_, 0, 3);
      }
    }
#ifdef USE_ASYNC_LDS
    __builtin_amdgcn_s_wait_asynccnt(0);
#endif
    __syncthreads();

    // scores S = Q(16x64) x K^T: two 16x16 tiles (keys j..j+15, j+16..j+31)
    v8f s[2] = {};
#pragma unroll
    for (int kt = 0; kt < 2; ++kt) {
#pragma unroll
      for (int frag = 0; frag < 2; ++frag) {
        const unsigned short* kp =
            &lds_k[(kt * 16 + ln) * 64 + half * 8 + frag * 32];
        const v16bf bk = frag_ld(kp, kp + 16);
        s[kt] = __builtin_amdgcn_wmma_f32_16x16x32_bf16(false, aq[frag], false,
                                                        bk, (short)0, s[kt],
                                                        false, false);
      }
    }

    // causal mask + online softmax (row = r + 8*half, key col = lane%16)
    float p0[8], p1[8];
#pragma unroll
    for (int r = 0; r < 8; ++r) {
      const int m = rowbase + r + half * 8;
      float x0 = s[0][r] * 0.125f;          // 1/sqrt(64)
      float x1 = s[1][r] * 0.125f;
      if (j + ln > m)      x0 = -1e30f;
      if (j + 16 + ln > m) x1 = -1e30f;
      float mx = fmaxf(x0, x1);
#pragma unroll
      for (int d = 1; d < 16; d <<= 1) mx = fmaxf(mx, __shfl_xor(mx, d, 16));
      const float mnew  = fmaxf(mrow[r], mx);
      const float alpha = __expf(mrow[r] - mnew);
      mrow[r] = mnew;
      x0 = __expf(x0 - mnew);
      x1 = __expf(x1 - mnew);
      float rs = x0 + x1;
#pragma unroll
      for (int d = 1; d < 16; d <<= 1) rs += __shfl_xor(rs, d, 16);
      lrow[r] = lrow[r] * alpha + rs;
      p0[r] = x0; p1[r] = x1;
#pragma unroll
      for (int ct = 0; ct < 4; ++ct) o[ct][r] *= alpha;
    }

    // P (C-layout f32) -> bf16 A-fragment layout via per-wave LDS transpose
#pragma unroll
    for (int r = 0; r < 8; ++r) {
      const int m = r + half * 8;
      lds_p[wave][m * 32 + ln]      = f2bfu(p0[r]);
      lds_p[wave][m * 32 + 16 + ln] = f2bfu(p1[r]);
    }
    __syncthreads();
    const unsigned short* pp = &lds_p[wave][ln * 32 + half * 8];
    const v16bf ap = frag_ld(pp, pp + 16);

    // O += P(16x32) x V(32x64): 4 head-dim tiles; B from transposed V tile
#pragma unroll
    for (int ct = 0; ct < 4; ++ct) {
      const unsigned short* vp = &lds_vt[(ct * 16 + ln) * 32 + half * 8];
      const v16bf bv = frag_ld(vp, vp + 16);
      o[ct] = __builtin_amdgcn_wmma_f32_16x16x32_bf16(false, ap, false, bv,
                                                      (short)0, o[ct], false,
                                                      false);
    }
  }

  // normalize and store f32 output
#pragma unroll
  for (int r = 0; r < 8; ++r) {
    const int m   = rowbase + r + half * 8;
    const float inv = 1.0f / lrow[r];
    float* orow = out + ((size_t)b * S_ + m) * DH_;
#pragma unroll
    for (int ct = 0; ct < 4; ++ct) orow[ct * 16 + ln] = o[ct][r] * inv;
  }
}

extern "C" void kernel_launch(void* const* d_in, const int* in_sizes, int n_in,
                              void* d_out, int out_size, void* d_ws, size_t ws_size,
                              hipStream_t stream) {
  (void)in_sizes; (void)n_in; (void)out_size; (void)ws_size;
  const float* q  = (const float*)d_in[0];
  const float* k  = (const float*)d_in[1];
  const float* v  = (const float*)d_in[2];
  const float* Wq = (const float*)d_in[3];
  const float* Wk = (const float*)d_in[4];
  const float* Wv = (const float*)d_in[5];

  unsigned short* qh  = (unsigned short*)d_ws;            // 2 MB each
  unsigned short* kh  = qh  + (size_t)B_ * S_ * DH_;
  unsigned short* vh  = kh  + (size_t)B_ * S_ * DH_;
  unsigned short* wtq = vh  + (size_t)B_ * S_ * DH_;      // 128 KB each
  unsigned short* wtk = wtq + (size_t)DH_ * DM_;
  unsigned short* wtv = wtk + (size_t)DH_ * DM_;

  wt_kernel<<<dim3((DH_ * DM_) / 256, 3), 256, 0, stream>>>(Wq, Wk, Wv,
                                                            wtq, wtk, wtv);
  proj_kernel<<<dim3((B_ * S_) / 16, 3), 128, 0, stream>>>(q, k, v, wtq, wtk,
                                                           wtv, qh, kh, vh);
  attn_kernel<<<dim3((B_ * S_) / 64), 128, 0, stream>>>(qh, kh, vh,
                                                        (float*)d_out);
}
